// HeteroGAT_8856222565026
// MI455X (gfx1250) — compile-verified, matrix operand order checked by reading, not run on hardware
//
#include <hip/hip_runtime.h>
#include <hip/hip_bf16.h>
#include <math.h>

typedef __attribute__((ext_vector_type(16))) __bf16 v16bf;
typedef __attribute__((ext_vector_type(8)))  __bf16 v8bf;
typedef __attribute__((ext_vector_type(8)))  float  v8f;

#define NUSERS 100000
#define NPROD  50000
#define NEDGE  250000
#define EMB    128
#define HID    256
#define NHEAD  4
#define HC     64

// ---------------- helpers ----------------
__device__ __forceinline__ unsigned f_ord(float x) {
  unsigned b = __float_as_uint(x);
  return (b & 0x80000000u) ? ~b : (b | 0x80000000u);
}
__device__ __forceinline__ float f_unord(unsigned u) {
  unsigned b = (u & 0x80000000u) ? (u ^ 0x80000000u) : ~u;
  return __uint_as_float(b);
}

// ---------------- weight packer: W[K,M] f32 -> WMMA B-fragment bf16 ----------
// Layout: [kk32][tile][lane][16 elems]; per lane 32 contiguous bytes.
__global__ void pack_w_k(const float* __restrict__ W, __bf16* __restrict__ out,
                         int K, int M) {
  int i = blockIdx.x * blockDim.x + threadIdx.x;
  int total = (K >> 5) * (M >> 4) * 32;
  if (i >= total) return;
  int lane = i & 31;
  int rest = i >> 5;
  int tilesM = M >> 4;
  int kk32 = rest / tilesM;
  int tAbs = rest - kk32 * tilesM;
  int n = (lane & 15) + (tAbs << 4);
  int half = lane >> 4;
  __bf16* o = out + (size_t)i * 16;
#pragma unroll
  for (int j = 0; j < 8; ++j) {
    int k0 = ((j & 4) ? 16 : 0) + half * 8 + (j & 3) * 2 + (kk32 << 5);
    o[2 * j]     = (__bf16)W[(size_t)k0 * M + n];
    o[2 * j + 1] = (__bf16)W[(size_t)(k0 + 1) * M + n];
  }
}

// ---------------- WMMA GEMM: C[N,M] = epi(A[N,K] @ W[K,M] + bias) -----------
// A: bf16 row-major. Wp: pre-packed bf16 fragments. f32 accumulate.
// Each wave32 computes a 16x128 strip (8 x 16x16 tiles); A frag loaded once
// per k-step as two b128 loads, each B frag one 32B load.
// mode: 0 = store f32, 1 = accumulate f32, 2 = relu f32, 3 = relu -> bf16 Cb.
__global__ __launch_bounds__(256)
void gemm_wmma_bf16(const __bf16* __restrict__ A, const __bf16* __restrict__ Wp,
                    const float* __restrict__ bias, float* __restrict__ C,
                    __bf16* __restrict__ Cb, int N, int K, int M, int mode) {
  const int lane = threadIdx.x & 31;
  const int wave = threadIdx.x >> 5;
  const int strip = blockIdx.x * (blockDim.x >> 5) + wave;
  const int colStrips = M >> 7;                  // M/128 (1 or 2)
  const int rowTile = strip / colStrips;
  const int colStrip = strip - rowTile * colStrips;
  const int m0 = rowTile << 4;
  if (m0 >= N) return;                           // wave-uniform: EXEC all-1s
  const int n0 = colStrip << 7;
  const int tilesM = M >> 4;
  const int t0 = n0 >> 4;
  const int half = lane >> 4;
  const int l15  = lane & 15;

  v8f acc[8];
#pragma unroll
  for (int t = 0; t < 8; ++t) {
    v8f z = {0.f,0.f,0.f,0.f,0.f,0.f,0.f,0.f};
    acc[t] = z;
  }

  const __bf16* Arow = A + (size_t)(m0 + l15) * K + half * 8;
  const int kSteps = K >> 5;
  for (int kk32 = 0; kk32 < kSteps; ++kk32) {
    const __bf16* ar = Arow + (kk32 << 5);
    if (kk32 + 1 < kSteps) __builtin_prefetch(ar + 32, 0, 1);
    v8bf lo = *(const v8bf*)ar;            // k = kk+half*8 .. +7
    v8bf hi = *(const v8bf*)(ar + 16);     // k = kk+16+half*8 .. +7
    v16bf af;
#pragma unroll
    for (int j = 0; j < 8; ++j) { af[j] = lo[j]; af[8 + j] = hi[j]; }
    const __bf16* bbase = Wp + (((size_t)kk32 * tilesM + t0) * 32 + lane) * 16;
#pragma unroll
    for (int t = 0; t < 8; ++t) {
      v16bf bfr = *(const v16bf*)(bbase + (size_t)t * 32 * 16);
      acc[t] = __builtin_amdgcn_wmma_f32_16x16x32_bf16(
          false, af, false, bfr, (short)0, acc[t], false, false);
    }
  }
  // epilogue: D layout — lane = col within tile, VGPR i = row i + 8*half
#pragma unroll
  for (int t = 0; t < 8; ++t) {
    int col = n0 + t * 16 + l15;
    float b = bias ? bias[col] : 0.f;
#pragma unroll
    for (int i = 0; i < 8; ++i) {
      int row = m0 + i + 8 * half;
      float v = acc[t][i] + b;
      size_t idx = (size_t)row * M + col;
      if (mode == 0)      C[idx]  = v;
      else if (mode == 1) C[idx] += v;
      else if (mode == 2) C[idx]  = v > 0.f ? v : 0.f;
      else                Cb[idx] = (__bf16)(v > 0.f ? v : 0.f);
    }
  }
}

// ---------------- small kernels ----------------
__global__ void cvt_bf16_k(const float* __restrict__ in, __bf16* __restrict__ out, int n) {
  int i = blockIdx.x * blockDim.x + threadIdx.x;
  if (i < n) out[i] = (__bf16)in[i];
}

__global__ void feat_k(const int* __restrict__ cid, const float* __restrict__ xnum,
                       const float* __restrict__ cemb, const float* __restrict__ Wf,
                       const float* __restrict__ bf, float* __restrict__ out,
                       __bf16* __restrict__ outb) {
  int i = blockIdx.x * blockDim.x + threadIdx.x;
  if (i >= NPROD * 128) return;
  int n = i >> 7, j = i & 127;
  const float* ce = cemb + (size_t)cid[n] * 32;
  float s = bf[j];
#pragma unroll
  for (int k = 0; k < 32; ++k) s += ce[k] * Wf[k * 128 + j];
  const float* xn = xnum + (size_t)n * 3;
  s += xn[0] * Wf[32 * 128 + j] + xn[1] * Wf[33 * 128 + j] + xn[2] * Wf[34 * 128 + j];
  s = s > 0.f ? s : 0.f;
  out[i] = s;
  outb[i] = (__bf16)s;
}

__global__ void gate_fuse_k(const float* __restrict__ tmp, const float* __restrict__ pe,
                            const float* __restrict__ ft, __bf16* __restrict__ xpb) {
  int i = blockIdx.x * blockDim.x + threadIdx.x;
  if (i >= NPROD * 128) return;
  float g = 1.f / (1.f + __expf(-tmp[i]));
  xpb[i] = (__bf16)(g * pe[i] + (1.f - g) * ft[i]);
}

__global__ void att_reduce_k(const float* __restrict__ X, const float* __restrict__ att,
                             float* __restrict__ out, int N) {
  int i = blockIdx.x * blockDim.x + threadIdx.x;
  if (i >= N * NHEAD) return;
  int n = i >> 2, h = i & 3;
  const float* x = X + (size_t)n * HID + h * HC;
  const float* a = att + h * HC;
  float s = 0.f;
#pragma unroll
  for (int c = 0; c < HC; ++c) s += x[c] * a[c];
  out[i] = s;
}

__global__ void ce_k(const float* __restrict__ We, const float* __restrict__ ae,
                     float* __restrict__ ce) {
  __shared__ float red[256];
  int t = threadIdx.x, h = t >> 6, c = t & 63;
  red[t] = We[h * 64 + c] * ae[h * 64 + c];
  __syncthreads();
  for (int s = 32; s > 0; s >>= 1) {
    if (c < s) red[t] += red[t + s];
    __syncthreads();
  }
  if (c == 0) ce[h] = red[t];
}

__global__ void bias_init_k(float* __restrict__ out, const float* __restrict__ bias, int n) {
  int i = blockIdx.x * blockDim.x + threadIdx.x;
  if (i < n) out[i] = bias[i & 255];
}

__global__ void seg_init_k(unsigned* __restrict__ mx, float* __restrict__ den, int n) {
  int i = blockIdx.x * blockDim.x + threadIdx.x;
  if (i < n) { mx[i] = 0u; den[i] = 0.f; }
}

__global__ void edge_alpha_k(const int* __restrict__ src, const int* __restrict__ dst,
                             const float* __restrict__ w, const float* __restrict__ asrc,
                             const float* __restrict__ adst, const float* __restrict__ ce,
                             float* __restrict__ alpha, unsigned* __restrict__ mx) {
  int e = blockIdx.x * blockDim.x + threadIdx.x;
  if (e >= NEDGE) return;
  int s = src[e], d = dst[e];
  float we = w[e];
#pragma unroll
  for (int h = 0; h < NHEAD; ++h) {
    float a = asrc[s * 4 + h] + adst[d * 4 + h] + we * ce[h];
    a = a > 0.f ? a : 0.2f * a;              // leaky_relu 0.2
    alpha[e * 4 + h] = a;
    atomicMax(&mx[d * 4 + h], f_ord(a));     // segment max (ordered-uint)
  }
}

__global__ void edge_exp_k(const int* __restrict__ dst, float* __restrict__ alpha,
                           const unsigned* __restrict__ mx, float* __restrict__ den) {
  int e = blockIdx.x * blockDim.x + threadIdx.x;
  if (e >= NEDGE) return;
  int d = dst[e];
#pragma unroll
  for (int h = 0; h < NHEAD; ++h) {
    float m = f_unord(mx[d * 4 + h]);
    float ex = __expf(alpha[e * 4 + h] - m);
    alpha[e * 4 + h] = ex;
    atomicAdd(&den[d * 4 + h], ex);
  }
}

__global__ void edge_coef_k(const int* __restrict__ dst, float* __restrict__ alpha,
                            const float* __restrict__ den) {
  int i = blockIdx.x * blockDim.x + threadIdx.x;
  if (i >= NEDGE * NHEAD) return;
  int e = i >> 2, h = i & 3;
  alpha[i] = alpha[i] / (den[dst[e] * 4 + h] + 1e-16f);
}

__global__ void edge_scatter_k(const int* __restrict__ src, const int* __restrict__ dst,
                               const float* __restrict__ coef, const float* __restrict__ xs,
                               float* __restrict__ out) {
  int i = blockIdx.x * blockDim.x + threadIdx.x;
  if (i >= NEDGE * 64) return;
  int e = i >> 6, c = i & 63;
  int s = src[e], d = dst[e];
  const float* xsr = xs + (size_t)s * HID;
  float* op = out + (size_t)d * HID;
#pragma unroll
  for (int h = 0; h < NHEAD; ++h) {
    atomicAdd(&op[h * HC + c], xsr[h * HC + c] * coef[e * 4 + h]);
  }
}

__global__ void elu_bf16_k(const float* __restrict__ c, __bf16* __restrict__ x, int n) {
  int i = blockIdx.x * blockDim.x + threadIdx.x;
  if (i < n) {
    float v = c[i];
    x[i] = (__bf16)(v > 0.f ? v : (__expf(v) - 1.f));
  }
}

__global__ void l2norm_rows_k(float* __restrict__ X, int N, int M) {
  int row = blockIdx.x * (blockDim.x >> 5) + (threadIdx.x >> 5);
  if (row >= N) return;
  int lane = threadIdx.x & 31;
  float* p = X + (size_t)row * M;
  float ss = 0.f;
  for (int j = lane; j < M; j += 32) { float v = p[j]; ss += v * v; }
  for (int m = 16; m > 0; m >>= 1) ss += __shfl_xor(ss, m, 32);
  float d = fmaxf(sqrtf(ss), 1e-12f);
  float inv = 1.f / d;
  for (int j = lane; j < M; j += 32) p[j] *= inv;
}

// ---------------- host orchestration ----------------
struct Gat { const float *W, *att_src, *att_dst, *W_e, *att_edge, *bias; };

static inline int cdiv(long long a, int b) { return (int)((a + b - 1) / b); }

extern "C" void kernel_launch(void* const* d_in, const int* in_sizes, int n_in,
                              void* d_out, int out_size, void* d_ws, size_t ws_size,
                              hipStream_t stream) {
  (void)in_sizes; (void)n_in; (void)out_size; (void)ws_size;
  // ---- inputs (setup_inputs dict order; params flattened in insertion order) ----
  const int*   category_id = (const int*)  d_in[0];
  const float* x_num       = (const float*)d_in[1];
  const int*   src_up      = (const int*)  d_in[2];
  const int*   dst_up      = (const int*)  d_in[3];
  const float* w_up        = (const float*)d_in[4];
  const int*   src_pu      = (const int*)  d_in[5];
  const int*   dst_pu      = (const int*)  d_in[6];
  const float* w_pu        = (const float*)d_in[7];
  const float* user_emb    = (const float*)d_in[8];
  const float* prod_emb    = (const float*)d_in[9];
  const float* cat_emb     = (const float*)d_in[10];
  const float* feat_W = (const float*)d_in[11];
  const float* feat_b = (const float*)d_in[12];
  const float* gate_W = (const float*)d_in[13];
  const float* gate_b = (const float*)d_in[14];
  auto gat = [&](int base) {
    Gat g;
    g.W        = (const float*)d_in[base + 0];
    g.att_src  = (const float*)d_in[base + 1];
    g.att_dst  = (const float*)d_in[base + 2];
    g.W_e      = (const float*)d_in[base + 3];
    g.att_edge = (const float*)d_in[base + 4];
    g.bias     = (const float*)d_in[base + 5];
    return g;
  };
  Gat c1up = gat(15), c1pu = gat(21), c2up = gat(27), c2pu = gat(33);
  const float* s1u_W = (const float*)d_in[39]; const float* s1u_b = (const float*)d_in[40];
  const float* s1p_W = (const float*)d_in[41]; const float* s1p_b = (const float*)d_in[42];
  const float* s2u_W = (const float*)d_in[43]; const float* s2u_b = (const float*)d_in[44];
  const float* s2p_W = (const float*)d_in[45]; const float* s2p_b = (const float*)d_in[46];
  const float* up1_W = (const float*)d_in[47]; const float* up1_b = (const float*)d_in[48];
  const float* up2_W = (const float*)d_in[49]; const float* up2_b = (const float*)d_in[50];
  const float* pp1_W = (const float*)d_in[51]; const float* pp1_b = (const float*)d_in[52];
  const float* pp2_W = (const float*)d_in[53]; const float* pp2_b = (const float*)d_in[54];

  // ---- workspace carve-out (256B aligned regions) ----
  char* base = (char*)d_ws;
  size_t off = 0;
  auto alloc = [&](size_t bytes) -> void* {
    void* p = base + off;
    off += (bytes + 255) & ~(size_t)255;
    return p;
  };
  float* XS   = (float*)alloc((size_t)NUSERS * HID * 4);   // src-transformed (f32)
  float* XD   = (float*)alloc((size_t)NUSERS * HID * 4);   // dst-transformed (f32)
  float* CU   = (float*)alloc((size_t)NUSERS * HID * 4);   // conv accum users
  float* CP   = (float*)alloc((size_t)NPROD  * HID * 4);   // conv accum products
  float* FEAT = (float*)alloc((size_t)NPROD  * EMB * 4);
  float*    ASRC = (float*)   alloc((size_t)NUSERS * NHEAD * 4);
  float*    ADST = (float*)   alloc((size_t)NUSERS * NHEAD * 4);
  unsigned* MX   = (unsigned*)alloc((size_t)NUSERS * NHEAD * 4);
  float*    DEN  = (float*)   alloc((size_t)NUSERS * NHEAD * 4);
  float*    ALPH = (float*)   alloc((size_t)NEDGE * NHEAD * 4);
  float*    CE   = (float*)   alloc(4 * 4);
  __bf16* XUB   = (__bf16*)alloc((size_t)NUSERS * HID * 2);  // user acts (bf16)
  __bf16* XPB   = (__bf16*)alloc((size_t)NPROD  * HID * 2);  // product acts (bf16)
  __bf16* UEB   = (__bf16*)alloc((size_t)NUSERS * EMB * 2);  // bf16 user_emb
  __bf16* PEB   = (__bf16*)alloc((size_t)NPROD  * EMB * 2);  // bf16 prod_emb
  __bf16* FEATB = (__bf16*)alloc((size_t)NPROD  * EMB * 2);
  __bf16* TB    = (__bf16*)alloc((size_t)NUSERS * HID * 2);  // tower tmp (bf16)
  __bf16* WPK   = (__bf16*)alloc((size_t)HID * HID * 2);     // packed weight scratch

  float* outU = (float*)d_out;
  float* outP = outU + (size_t)NUSERS * EMB;

  auto gemm = [&](const __bf16* A, const float* W, const float* bias, float* Cm,
                  __bf16* Cb, int N, int K, int M, int mode) {
    int packN = (K >> 5) * (M >> 4) * 32;
    hipLaunchKernelGGL(pack_w_k, dim3(cdiv(packN, 256)), dim3(256), 0, stream,
                       W, WPK, K, M);
    int strips = (N / 16) * (M / 128);
    hipLaunchKernelGGL(gemm_wmma_bf16, dim3(cdiv(strips, 8)), dim3(256), 0, stream,
                       A, WPK, bias, Cm, Cb, N, K, M, mode);
  };

  auto run_conv = [&](const __bf16* xsrc, int Ns, const __bf16* xdst, int Nd, int Kl,
                      const int* src, const int* dst, const float* ew,
                      const Gat& g, float* OUT) {
    gemm(xsrc, g.W, nullptr, XS, nullptr, Ns, Kl, HID, 0);
    gemm(xdst, g.W, nullptr, XD, nullptr, Nd, Kl, HID, 0);
    hipLaunchKernelGGL(att_reduce_k, dim3(cdiv((long long)Ns * NHEAD, 256)), dim3(256), 0, stream, XS, g.att_src, ASRC, Ns);
    hipLaunchKernelGGL(att_reduce_k, dim3(cdiv((long long)Nd * NHEAD, 256)), dim3(256), 0, stream, XD, g.att_dst, ADST, Nd);
    hipLaunchKernelGGL(ce_k, dim3(1), dim3(256), 0, stream, g.W_e, g.att_edge, CE);
    hipLaunchKernelGGL(bias_init_k, dim3(cdiv((long long)Nd * HID, 256)), dim3(256), 0, stream, OUT, g.bias, Nd * HID);
    hipLaunchKernelGGL(seg_init_k, dim3(cdiv((long long)Nd * NHEAD, 256)), dim3(256), 0, stream, MX, DEN, Nd * NHEAD);
    hipLaunchKernelGGL(edge_alpha_k, dim3(cdiv(NEDGE, 256)), dim3(256), 0, stream, src, dst, ew, ASRC, ADST, CE, ALPH, MX);
    hipLaunchKernelGGL(edge_exp_k, dim3(cdiv(NEDGE, 256)), dim3(256), 0, stream, dst, ALPH, MX, DEN);
    hipLaunchKernelGGL(edge_coef_k, dim3(cdiv((long long)NEDGE * NHEAD, 256)), dim3(256), 0, stream, dst, ALPH, DEN);
    hipLaunchKernelGGL(edge_scatter_k, dim3(cdiv((long long)NEDGE * 64, 256)), dim3(256), 0, stream, src, dst, ALPH, XS, OUT);
  };

  auto run_layer = [&](const __bf16* xu_in, const __bf16* xp_in, int Kl,
                       const Gat& gup, const Gat& gpu,
                       const float* suW, const float* sub,
                       const float* spW, const float* spb) {
    run_conv(xu_in, NUSERS, xp_in, NPROD, Kl, src_up, dst_up, w_up, gup, CP); // -> products
    run_conv(xp_in, NPROD, xu_in, NUSERS, Kl, src_pu, dst_pu, w_pu, gpu, CU); // -> users
    gemm(xu_in, suW, sub, CU, nullptr, NUSERS, Kl, HID, 1);   // skip accumulate
    gemm(xp_in, spW, spb, CP, nullptr, NPROD,  Kl, HID, 1);
    hipLaunchKernelGGL(elu_bf16_k, dim3(cdiv((long long)NUSERS * HID, 256)), dim3(256), 0, stream, CU, XUB, NUSERS * HID);
    hipLaunchKernelGGL(elu_bf16_k, dim3(cdiv((long long)NPROD * HID, 256)), dim3(256), 0, stream, CP, XPB, NPROD * HID);
  };

  // ---- stage 0: bf16 copies of embeddings + product feature fusion ----
  hipLaunchKernelGGL(cvt_bf16_k, dim3(cdiv((long long)NUSERS * EMB, 256)), dim3(256), 0, stream, user_emb, UEB, NUSERS * EMB);
  hipLaunchKernelGGL(cvt_bf16_k, dim3(cdiv((long long)NPROD * EMB, 256)), dim3(256), 0, stream, prod_emb, PEB, NPROD * EMB);
  hipLaunchKernelGGL(feat_k, dim3(cdiv((long long)NPROD * 128, 256)), dim3(256), 0, stream,
                     category_id, x_num, cat_emb, feat_W, feat_b, FEAT, FEATB);
  // gate logits: prod_emb @ gate_W[:128] + FEAT @ gate_W[128:] + gate_b (in CP as [NP,128])
  gemm(PEB,   gate_W,             gate_b,  CP, nullptr, NPROD, EMB, 128, 0);
  gemm(FEATB, gate_W + 128 * 128, nullptr, CP, nullptr, NPROD, EMB, 128, 1);
  hipLaunchKernelGGL(gate_fuse_k, dim3(cdiv((long long)NPROD * 128, 256)), dim3(256), 0, stream,
                     CP, prod_emb, FEAT, XPB);

  // ---- layers ----
  run_layer(UEB, XPB, EMB, c1up, c1pu, s1u_W, s1u_b, s1p_W, s1p_b);
  run_layer(XUB, XPB, HID, c2up, c2pu, s2u_W, s2u_b, s2p_W, s2p_b);

  // ---- towers + L2 normalize ----
  gemm(XUB, up1_W, up1_b, nullptr, TB, NUSERS, HID, HID, 3);   // relu -> bf16
  gemm(TB,  up2_W, up2_b, outU, nullptr, NUSERS, HID, EMB, 0);
  hipLaunchKernelGGL(l2norm_rows_k, dim3(cdiv(NUSERS, 8)), dim3(256), 0, stream, outU, NUSERS, EMB);
  gemm(XPB, pp1_W, pp1_b, nullptr, TB, NPROD, HID, HID, 3);    // relu -> bf16
  gemm(TB,  pp2_W, pp2_b, outP, nullptr, NPROD, HID, EMB, 0);
  hipLaunchKernelGGL(l2norm_rows_k, dim3(cdiv(NPROD, 8)), dim3(256), 0, stream, outP, NPROD, EMB);
}
